// GNNVirtualNode_77008763617706
// MI455X (gfx1250) — compile-verified, matrix operand order checked by reading, not run on hardware
//
#include <hip/hip_runtime.h>
#include <hip/hip_bf16.h>

typedef __attribute__((ext_vector_type(16))) _Float16 v16h;
typedef __attribute__((ext_vector_type(8)))  _Float16 v8h;
typedef __attribute__((ext_vector_type(8)))  float    v8f;

constexpr int Nn  = 50000;
constexpr int Ee  = 800000;
constexpr int Gg  = 256;
constexpr int DIN = 128;
constexpr int Hh  = 256;
constexpr int Cc  = 10;
constexpr float EPSV = 1e-5f;

// ---------------- utility kernels ----------------

__global__ void k_zero(float* __restrict__ p, int n) {
    int i = blockIdx.x * blockDim.x + threadIdx.x;
    if (i < n) p[i] = 0.0f;
}

__global__ void k_degree(const int* __restrict__ dst, float* __restrict__ cnt, int E) {
    int i = blockIdx.x * blockDim.x + threadIdx.x;
    if (i < E) atomicAdd(&cnt[dst[i]], 1.0f);
}

__global__ void k_inv(const float* __restrict__ cnt, float* __restrict__ rinv, int n) {
    int i = blockIdx.x * blockDim.x + threadIdx.x;
    if (i < n) rinv[i] = 1.0f / fmaxf(cnt[i], 1.0f);
}

// scatter-add of source features into agg[dst]  (blockDim.x == F)
__global__ void k_scatter(const float* __restrict__ x, const int* __restrict__ src,
                          const int* __restrict__ dst, float* __restrict__ agg,
                          int E, int F) {
    int f  = threadIdx.x;
    int e0 = blockIdx.x * 4;
#pragma unroll
    for (int k = 0; k < 4; ++k) {
        int e = e0 + k;
        if (e < E) {
            atomicAdd(&agg[(size_t)dst[e] * F + f], x[(size_t)src[e] * F + f]);
        }
    }
}

// W:[Fin,H] row-major fp32  ->  WT:[H,Fin] f16 (column-major of W)
__global__ void k_packW(const float* __restrict__ W, _Float16* __restrict__ WT,
                        int Fin, int Hc) {
    int i = blockIdx.x * blockDim.x + threadIdx.x;
    if (i < Fin * Hc) {
        int k = i / Hc;
        int c = i % Hc;
        WT[(size_t)c * Fin + k] = (_Float16)W[i];
    }
}

// ---------------- fused SAGE GEMM (WMMA) ----------------
// out[N,H] = relu( (agg*rinv) @ Wl + x @ Wr + bl (+ x residual) )
// Each wave computes a 16x64 strip (4 column tiles): A operands are loaded
// and converted ONCE per K-step and reused across 8 WMMAs.
// Also accumulates per-column sum / sum-of-squares for BatchNorm.

#define CVT8(Avec, BASE, p, q, s)                                              \
    Avec[BASE + 0] = (_Float16)((p).x * (s));                                  \
    Avec[BASE + 1] = (_Float16)((p).y * (s));                                  \
    Avec[BASE + 2] = (_Float16)((p).z * (s));                                  \
    Avec[BASE + 3] = (_Float16)((p).w * (s));                                  \
    Avec[BASE + 4] = (_Float16)((q).x * (s));                                  \
    Avec[BASE + 5] = (_Float16)((q).y * (s));                                  \
    Avec[BASE + 6] = (_Float16)((q).z * (s));                                  \
    Avec[BASE + 7] = (_Float16)((q).w * (s));

__global__ __launch_bounds__(128)
void k_sage_gemm(const float* __restrict__ xin, const float* __restrict__ agg,
                 const float* __restrict__ rinv,
                 const _Float16* __restrict__ WlT, const _Float16* __restrict__ WrT,
                 const float* __restrict__ bl, float* __restrict__ out,
                 float* __restrict__ bnsum, float* __restrict__ bnss,
                 int Fin, int residual) {
    const int lane = threadIdx.x & 31;
    const int wave = threadIdx.x >> 5;        // column group 0..3 (64 cols each)
    const int tm   = blockIdx.x;              // row tile 0..N/16-1

    const int r15   = lane & 15;
    const int hi    = lane >> 4;              // 0: lanes 0-15, 1: lanes 16-31
    const int arow  = tm * 16 + r15;          // A-matrix row for this lane
    const int base0 = hi ? 8 : 0;             // K offsets per ISA 16-bit layout
    const int base1 = base0 + 16;

    const float rv = rinv[arow];
    const float* aggp = agg + (size_t)arow * Fin;
    const float* xp   = xin + (size_t)arow * Fin;

    const _Float16* wlp[4];
    const _Float16* wrp[4];
#pragma unroll
    for (int j = 0; j < 4; ++j) {
        int bcol = wave * 64 + j * 16 + r15;  // B-matrix column for this lane
        wlp[j] = WlT + (size_t)bcol * Fin;
        wrp[j] = WrT + (size_t)bcol * Fin;
    }

    v8f c[4];
#pragma unroll
    for (int j = 0; j < 4; ++j) c[j] = (v8f){};

    for (int k = 0; k < Fin; k += 32) {
        if (k + 32 < Fin) {
            __builtin_prefetch(aggp + k + 32, 0, 1);
            __builtin_prefetch(xp + k + 32, 0, 1);
        }
        v16h Am, Ax;
        {   // mean operand (agg * 1/deg)
            float4 a0 = *(const float4*)(aggp + k + base0);
            float4 a1 = *(const float4*)(aggp + k + base0 + 4);
            float4 a2 = *(const float4*)(aggp + k + base1);
            float4 a3 = *(const float4*)(aggp + k + base1 + 4);
            CVT8(Am, 0, a0, a1, rv)
            CVT8(Am, 8, a2, a3, rv)
        }
        {   // x operand
            float4 x0 = *(const float4*)(xp + k + base0);
            float4 x1 = *(const float4*)(xp + k + base0 + 4);
            float4 x2 = *(const float4*)(xp + k + base1);
            float4 x3 = *(const float4*)(xp + k + base1 + 4);
            CVT8(Ax, 0, x0, x1, 1.0f)
            CVT8(Ax, 8, x2, x3, 1.0f)
        }
#pragma unroll
        for (int j = 0; j < 4; ++j) {
            v16h B;
            v8h b0 = *(const v8h*)(wlp[j] + k + base0);
            v8h b1 = *(const v8h*)(wlp[j] + k + base1);
#pragma unroll
            for (int q = 0; q < 8; ++q) { B[q] = b0[q]; B[8 + q] = b1[q]; }
            c[j] = __builtin_amdgcn_wmma_f32_16x16x32_f16(false, Am, false, B,
                                                          (short)0, c[j], false, false);
            v8h r0 = *(const v8h*)(wrp[j] + k + base0);
            v8h r1 = *(const v8h*)(wrp[j] + k + base1);
#pragma unroll
            for (int q = 0; q < 8; ++q) { B[q] = r0[q]; B[8 + q] = r1[q]; }
            c[j] = __builtin_amdgcn_wmma_f32_16x16x32_f16(false, Ax, false, B,
                                                          (short)0, c[j], false, false);
        }
    }

    // epilogue: +bias (+residual), relu, store, BN stats
    const int rowbase = tm * 16 + (hi ? 8 : 0);   // C/D layout: VGPR r -> row rowbase+r
#pragma unroll
    for (int j = 0; j < 4; ++j) {
        const int bcol = wave * 64 + j * 16 + r15;
        const float bb = bl[bcol];
        float s = 0.0f, ss = 0.0f;
        if (residual) {
#pragma unroll
            for (int r = 0; r < 8; ++r) {
                float v = c[j][r] + bb + xin[(size_t)(rowbase + r) * Fin + bcol];
                v = fmaxf(v, 0.0f);
                out[(size_t)(rowbase + r) * Hh + bcol] = v;
                s += v; ss += v * v;
            }
        } else {
#pragma unroll
            for (int r = 0; r < 8; ++r) {
                float v = fmaxf(c[j][r] + bb, 0.0f);
                out[(size_t)(rowbase + r) * Hh + bcol] = v;
                s += v; ss += v * v;
            }
        }
        atomicAdd(&bnsum[bcol], s);
        atomicAdd(&bnss[bcol], ss);
    }
}

// ---------------- BatchNorm finish ----------------

__global__ void k_bnscale(const float* __restrict__ sum, const float* __restrict__ ss,
                          const float* __restrict__ g, const float* __restrict__ b,
                          float* __restrict__ scale, float* __restrict__ shift, int n) {
    int i = blockIdx.x * blockDim.x + threadIdx.x;
    if (i < n) {
        float mu  = sum[i] / (float)Nn;
        float var = fmaxf(ss[i] / (float)Nn - mu * mu, 0.0f);
        float sc  = g[i] * rsqrtf(var + EPSV);
        scale[i] = sc;
        shift[i] = b[i] - mu * sc;
    }
}

__global__ void k_bnapply(float* __restrict__ h, const float* __restrict__ scale,
                          const float* __restrict__ shift, int total) {
    for (int i = blockIdx.x * blockDim.x + threadIdx.x; i < total;
         i += gridDim.x * blockDim.x) {
        int col = i & (Hh - 1);
        h[i] = h[i] * scale[col] + shift[col];
    }
}

// ---------------- virtual-node readout ----------------

__global__ void k_lastidx(const int* __restrict__ batch, int* __restrict__ last, int n) {
    int i = blockIdx.x * blockDim.x + threadIdx.x;
    if (i < n) {
        int b = batch[i];
        if (i == n - 1 || batch[i + 1] != b) last[b] = i;
    }
}

__global__ void k_readout(const float* __restrict__ h, const int* __restrict__ last,
                          const float* __restrict__ Wout, const float* __restrict__ bout,
                          float* __restrict__ out) {
    __shared__ float vs[Hh];
    int g = blockIdx.x;
    int t = threadIdx.x;              // blockDim = 32
    int li = last[g];
    for (int i = t; i < Hh; i += 32) vs[i] = h[(size_t)li * Hh + i];
    __syncthreads();
    if (t < Cc) {
        float acc = bout[t];
        for (int i = 0; i < Hh; ++i) acc += vs[i] * Wout[i * Cc + t];
        out[g * Cc + t] = acc;
    }
}

// ---------------- host launcher ----------------

extern "C" void kernel_launch(void* const* d_in, const int* in_sizes, int n_in,
                              void* d_out, int out_size, void* d_ws, size_t ws_size,
                              hipStream_t stream) {
    const float* x     = (const float*)d_in[0];
    const int*   eidx  = (const int*)d_in[1];
    const int*   batch = (const int*)d_in[2];
    // d_in[3] = num_graphs scalar (constant Gg here)
    const float* Wl[3] = {(const float*)d_in[4],  (const float*)d_in[9],  (const float*)d_in[14]};
    const float* bl[3] = {(const float*)d_in[5],  (const float*)d_in[10], (const float*)d_in[15]};
    const float* Wr[3] = {(const float*)d_in[6],  (const float*)d_in[11], (const float*)d_in[16]};
    const float* gg[3] = {(const float*)d_in[7],  (const float*)d_in[12], (const float*)d_in[17]};
    const float* bb[3] = {(const float*)d_in[8],  (const float*)d_in[13], (const float*)d_in[18]};
    const float* Wout  = (const float*)d_in[19];
    const float* bout  = (const float*)d_in[20];
    const int* src = eidx;
    const int* dst = eidx + Ee;

    // workspace carve-out
    char*  ws  = (char*)d_ws;
    size_t off = 0;
    auto carve = [&](size_t bytes) -> void* {
        void* p = ws + off;
        off = (off + bytes + 255) & ~(size_t)255;
        return p;
    };
    float* agg   = (float*)carve((size_t)Nn * Hh * 4);
    float* bufA  = (float*)carve((size_t)Nn * Hh * 4);
    float* bufB  = (float*)carve((size_t)Nn * Hh * 4);
    float* cnt   = (float*)carve((size_t)Nn * 4);
    float* rinv  = (float*)carve((size_t)Nn * 4);
    float* bnsum = (float*)carve(Hh * 4);
    float* bnss  = (float*)carve(Hh * 4);
    float* bnsc  = (float*)carve(Hh * 4);
    float* bnsh  = (float*)carve(Hh * 4);
    int*   last  = (int*)carve(Gg * 4);
    int fins[3] = {DIN, Hh, Hh};
    _Float16* WlT[3];
    _Float16* WrT[3];
    for (int i = 0; i < 3; ++i) {
        WlT[i] = (_Float16*)carve((size_t)Hh * fins[i] * 2);
        WrT[i] = (_Float16*)carve((size_t)Hh * fins[i] * 2);
    }

    // degrees and reciprocal counts (shared by all 3 blocks)
    k_zero<<<(Nn + 255) / 256, 256, 0, stream>>>(cnt, Nn);
    k_degree<<<(Ee + 255) / 256, 256, 0, stream>>>(dst, cnt, Ee);
    k_inv<<<(Nn + 255) / 256, 256, 0, stream>>>(cnt, rinv, Nn);

    // pack weights to f16, transposed for WMMA B-operand vector loads
    for (int i = 0; i < 3; ++i) {
        int tot = fins[i] * Hh;
        k_packW<<<(tot + 255) / 256, 256, 0, stream>>>(Wl[i], WlT[i], fins[i], Hh);
        k_packW<<<(tot + 255) / 256, 256, 0, stream>>>(Wr[i], WrT[i], fins[i], Hh);
    }

    const float* cur = x;
    float* outb[3] = {bufA, bufB, bufA};
    for (int i = 0; i < 3; ++i) {
        int F = fins[i];
        int aggElems = Nn * F;
        k_zero<<<(aggElems + 255) / 256, 256, 0, stream>>>(agg, aggElems);
        k_scatter<<<Ee / 4, F, 0, stream>>>(cur, src, dst, agg, Ee, F);
        k_zero<<<1, 256, 0, stream>>>(bnsum, Hh);
        k_zero<<<1, 256, 0, stream>>>(bnss, Hh);
        // one block per 16-row tile; 4 waves cover the 256 output columns
        k_sage_gemm<<<Nn / 16, 128, 0, stream>>>(cur, agg, rinv, WlT[i], WrT[i],
                                                 bl[i], outb[i], bnsum, bnss,
                                                 F, i > 0 ? 1 : 0);
        k_bnscale<<<1, 256, 0, stream>>>(bnsum, bnss, gg[i], bb[i], bnsc, bnsh, Hh);
        k_bnapply<<<1024, 256, 0, stream>>>(outb[i], bnsc, bnsh, Nn * Hh);
        cur = outb[i];
    }

    k_lastidx<<<(Nn + 255) / 256, 256, 0, stream>>>(batch, last, Nn);
    k_readout<<<Gg, 32, 0, stream>>>(cur, last, Wout, bout, (float*)d_out);
}
